// FastTransformer_82042465288570
// MI455X (gfx1250) — matrix operand order repaired and checked by reading, not compiled, not run on hardware
//
#include <hip/hip_runtime.h>
#include <hip/hip_bf16.h>

// ---------------------------------------------------------------------------
// Types for CDNA5 WMMA (wave32): v_wmma_f32_16x16x32_bf16
// ---------------------------------------------------------------------------
typedef __attribute__((ext_vector_type(16))) __bf16 v16bf;
typedef __attribute__((ext_vector_type(8)))  float  v8f;

#define WMMA_BF16(a, b, c) \
  __builtin_amdgcn_wmma_f32_16x16x32_bf16(false, (a), false, (b), (short)0, (c), false, false)

// Tensor Data Mover availability (device pass only; host pass falls back).
#if defined(__has_builtin)
# if __has_builtin(__builtin_amdgcn_tensor_load_to_lds) && \
     __has_builtin(__builtin_amdgcn_s_wait_tensorcnt)
#  define HAVE_TDM 1
# else
#  define HAVE_TDM 0
# endif
#else
# define HAVE_TDM 0
#endif

__device__ __forceinline__ unsigned short f2bf(float f) {
  union { float f; unsigned u; } c; c.f = f;
  unsigned u = c.u;
  u += 0x7fffu + ((u >> 16) & 1u);   // round-to-nearest-even
  return (unsigned short)(u >> 16);
}

#if HAVE_TDM
typedef __attribute__((ext_vector_type(4))) unsigned v4u_t;
typedef __attribute__((ext_vector_type(8))) int      v8i_t;
typedef __attribute__((ext_vector_type(4))) int      v4i_t;

// 2-D TDM load: tile_x (contiguous, elements) x tile_y rows, 2-byte elements,
// row stride stride_x elements, into LDS at byte offset lds_off.
// D# layout per CDNA5 ISA ch.8 (group0 + group1; groups 2/3 zero => 2D).
__device__ __forceinline__ void tdm_load_2d(unsigned lds_off, const void* g,
                                            unsigned tile_x, unsigned tile_y,
                                            unsigned stride_x) {
  size_t ga = (size_t)g;
  v4u_t g0;
  g0[0] = 1u;                                          // count=1 (valid user D#)
  g0[1] = lds_off;                                     // lds_addr
  g0[2] = (unsigned)ga;                                // global_addr[31:0]
  g0[3] = (unsigned)((ga >> 32) & 0x01ffffffu) | 0x80000000u;  // addr[56:32] | type=2
  v8i_t g1;
  g1[0] = 0x00010000;                                  // data_size=1 (2 bytes)
  g1[1] = 0;                                           // tensor_dim0 = 1<<22 (low16=0)
  g1[2] = 0x40;                                        // tensor_dim0 hi | tensor_dim1 lo
  g1[3] = (int)(0x40u | (tile_x << 16));               // tensor_dim1 hi | tile_dim0
  g1[4] = (int)(tile_y & 0xffffu);                     // tile_dim1 (tile_dim2=0)
  g1[5] = (int)stride_x;                               // tensor_dim0_stride lo32
  g1[6] = 0;
  g1[7] = 0;
  v4i_t z4 = {0, 0, 0, 0};
# if __clang_major__ >= 23
  v8i_t z8 = {0, 0, 0, 0, 0, 0, 0, 0};
  __builtin_amdgcn_tensor_load_to_lds(g0, g1, z4, z4, z8, 0);
# else
  __builtin_amdgcn_tensor_load_to_lds(g0, g1, z4, z4, 0);
# endif
}
#endif

// A-fragment: 16x32 bf16, rows = row0..row0+15, K = kbase..kbase+31.
// LDS row-major [row][k], stride ld (shorts). ISA 7.12.2 layout.
__device__ __forceinline__ v16bf frag_a(const unsigned short* p, int ld, int row0, int kbase) {
  const int l  = threadIdx.x & 31;
  const int m  = row0 + (l & 15);
  const int kh = (l >> 4) << 3;
  union { v16bf v; unsigned d[8]; } f;
  const unsigned short* rp = p + m * ld + kbase;
#pragma unroll
  for (int j = 0; j < 8; j++) {
    int k = ((j & 4) ? 16 : 0) + kh + ((j & 3) << 1);
    f.d[j] = *(const unsigned*)(rp + k);
  }
  return f.v;
}

// B-fragment: 32x16 bf16, cols = col0..col0+15, K = kbase..kbase+31.
// LDS layout TRANSPOSED: [n][k], stride ld, so each dword pair is one b32 read.
__device__ __forceinline__ v16bf frag_b(const unsigned short* p, int ld, int col0, int kbase) {
  const int l  = threadIdx.x & 31;
  const int n  = col0 + (l & 15);
  const int kh = (l >> 4) << 3;
  union { v16bf v; unsigned d[8]; } f;
  const unsigned short* cp = p + n * ld + kbase;
#pragma unroll
  for (int j = 0; j < 8; j++) f.d[j] = *(const unsigned*)(cp + ((j & 4) ? 16 : 0) + kh + ((j & 3) << 1));
  return f.v;
}

// ---------------------------------------------------------------------------
// GEMM: C[M,N] = A_bf16[M,K] @ W^T (Wt is bf16 [N,K]) + bias (+ReLU) (+resid)
// 256 threads (8 waves), tile 128(M) x 64(N), BK = 64, double-buffered LDS.
// TDM path: wave 0 issues tensor_load_to_lds for next stage, s_wait_tensorcnt.
// ---------------------------------------------------------------------------
template<bool RELU, bool RESID>
__device__ __forceinline__ void store_tile(float* C, const float* bias, const float* resid,
                                           int N, int gm, int gn, v8f acc) {
  float bb = bias[gn];
#pragma unroll
  for (int r = 0; r < 8; r++) {
    float v = acc[r] + bb;
    if (RELU)  v = fmaxf(v, 0.f);
    if (RESID) v += resid[(size_t)(gm + r) * N + gn];
    C[(size_t)(gm + r) * N + gn] = v;
  }
}

template<bool RELU, bool RESID>
__global__ __launch_bounds__(256) void gemm_wmma_kernel(
    const unsigned short* __restrict__ A,    // bf16 [M,K] row-major
    const unsigned short* __restrict__ Wt,   // bf16 [N,K] row-major (W transposed)
    const float* __restrict__ bias,          // [N]
    const float* __restrict__ resid,         // [M,N] or nullptr
    float* __restrict__ C,                   // [M,N]
    int M, int N, int K)
{
  __shared__ __attribute__((aligned(16))) unsigned short As[2][128 * 64];
  __shared__ __attribute__((aligned(16))) unsigned short Ws[2][64 * 64];

  const int tid  = threadIdx.x;
  const int lane = tid & 31;
  const int wid  = tid >> 5;
  const int wm   = wid >> 1;       // 0..3
  const int wn   = wid & 1;        // 0..1
  const int m0   = blockIdx.y * 128;
  const int n0   = blockIdx.x * 64;
  const int nstage = K >> 6;

  v8f acc00 = {}, acc01 = {}, acc10 = {}, acc11 = {};

  auto compute = [&](const unsigned short* as, const unsigned short* ws) {
    v16bf a00 = frag_a(as, 64, wm * 32 + 0,  0);
    v16bf a01 = frag_a(as, 64, wm * 32 + 0,  32);
    v16bf a10 = frag_a(as, 64, wm * 32 + 16, 0);
    v16bf a11 = frag_a(as, 64, wm * 32 + 16, 32);
    v16bf b00 = frag_b(ws, 64, wn * 32 + 0,  0);
    v16bf b01 = frag_b(ws, 64, wn * 32 + 0,  32);
    v16bf b10 = frag_b(ws, 64, wn * 32 + 16, 0);
    v16bf b11 = frag_b(ws, 64, wn * 32 + 16, 32);
    acc00 = WMMA_BF16(a00, b00, acc00);  acc00 = WMMA_BF16(a01, b01, acc00);
    acc01 = WMMA_BF16(a00, b10, acc01);  acc01 = WMMA_BF16(a01, b11, acc01);
    acc10 = WMMA_BF16(a10, b00, acc10);  acc10 = WMMA_BF16(a11, b01, acc10);
    acc11 = WMMA_BF16(a10, b10, acc11);  acc11 = WMMA_BF16(a11, b11, acc11);
  };

#if HAVE_TDM
  if (wid == 0) {
    tdm_load_2d((unsigned)(size_t)&As[0][0], A + (size_t)m0 * K, 64, 128, (unsigned)K);
    tdm_load_2d((unsigned)(size_t)&Ws[0][0], Wt + (size_t)n0 * K, 64, 64, (unsigned)K);
  }
  for (int s = 0; s < nstage; s++) {
    const int buf = s & 1;
    if (wid == 0) {
      if (s + 1 < nstage) {
        const int k0 = (s + 1) << 6;
        tdm_load_2d((unsigned)(size_t)&As[buf ^ 1][0], A + (size_t)m0 * K + k0, 64, 128, (unsigned)K);
        tdm_load_2d((unsigned)(size_t)&Ws[buf ^ 1][0], Wt + (size_t)n0 * K + k0, 64, 64, (unsigned)K);
        __builtin_amdgcn_s_wait_tensorcnt(2);   // stage s complete, s+1 in flight
      } else {
        __builtin_amdgcn_s_wait_tensorcnt(0);
      }
    }
    __syncthreads();
    compute(As[buf], Ws[buf]);
    __syncthreads();                            // all waves done before buf is reused
  }
#else
  const int arow = tid >> 1, ahalf = tid & 1;   // A: 2 threads/row, 64B each
  const int wrow = tid >> 2, wq = tid & 3;      // W: 4 threads/row, 32B each
  auto stage = [&](int s, int buf) {
    const int k0 = s << 6;
    const uint4* ga = (const uint4*)(A + (size_t)(m0 + arow) * K + k0 + ahalf * 32);
    uint4* da = (uint4*)(&As[buf][arow * 64 + ahalf * 32]);
    da[0] = ga[0]; da[1] = ga[1]; da[2] = ga[2]; da[3] = ga[3];
    const uint4* gw = (const uint4*)(Wt + (size_t)(n0 + wrow) * K + k0 + wq * 16);
    uint4* dw = (uint4*)(&Ws[buf][wrow * 64 + wq * 16]);
    dw[0] = gw[0]; dw[1] = gw[1];
  };
  stage(0, 0);
  for (int s = 0; s < nstage; s++) {
    const int buf = s & 1;
    __syncthreads();
    if (s + 1 < nstage) stage(s + 1, buf ^ 1);  // fill other buffer while computing
    compute(As[buf], Ws[buf]);
  }
#endif

  const int rbase = (lane >> 4) << 3;
  const int cl    = lane & 15;
  const int bm    = m0 + wm * 32 + rbase;
  const int bn    = n0 + wn * 32 + cl;
  store_tile<RELU, RESID>(C, bias, resid, N, bm,      bn,      acc00);
  store_tile<RELU, RESID>(C, bias, resid, N, bm,      bn + 16, acc01);
  store_tile<RELU, RESID>(C, bias, resid, N, bm + 16, bn,      acc10);
  store_tile<RELU, RESID>(C, bias, resid, N, bm + 16, bn + 16, acc11);
}

// ---------------------------------------------------------------------------
// FAVOR+ positive features. One wave per (token, head); lane = random feature.
// ---------------------------------------------------------------------------
__global__ __launch_bounds__(256) void favor_kernel(
    const float* __restrict__ x,      // [tokens, 512] (H=8, Dh=64)
    const float* __restrict__ omega,  // [64, 32]
    float* __restrict__ out,          // [tokens, 512] (H=8, M=64)
    int tokens)
{
  int gw    = blockIdx.x * 8 + (threadIdx.x >> 5);
  int lane  = threadIdx.x & 31;
  int token = gw >> 3;
  int head  = gw & 7;
  if (token >= tokens) return;

  const float s = 0.35355339059327379f;  // 64^-0.25
  const float* xp = x + (size_t)token * 512 + head * 64;
  float xa = xp[lane] * s;
  float xb = xp[lane + 32] * s;

  float ns = xa * xa + xb * xb;
#pragma unroll
  for (int o = 16; o > 0; o >>= 1) ns += __shfl_xor(ns, o, 32);

  float u = 0.f;
#pragma unroll 8
  for (int d = 0; d < 32; d++) u = fmaf(__shfl(xa, d, 32), omega[d * 32 + lane], u);
#pragma unroll 8
  for (int d = 0; d < 32; d++) u = fmaf(__shfl(xb, d, 32), omega[(d + 32) * 32 + lane], u);

  float off = 0.5f * ns;
  float* op = out + (size_t)token * 512 + head * 64;
  op[lane]      = __expf( u - off) * 0.125f;
  op[lane + 32] = __expf(-u - off) * 0.125f;
}

// ---------------------------------------------------------------------------
// Chunked causal linear attention. One block per (batch, head), chunk C=32.
//   A      = Qc Kc^T (masked)            2 WMMA x 4 waves
//   num    = A_masked Vc + Qc S_prev     3 WMMA x 8 waves
//   S     += Kc^T Vc                     2 WMMA x 8 waves
// ---------------------------------------------------------------------------
__global__ __launch_bounds__(256) void causal_attn_kernel(
    const float* __restrict__ Qf, const float* __restrict__ Kf,
    const float* __restrict__ V, float* __restrict__ Out, int L)
{
  __shared__ float          S[64][64];     // f32 state
  __shared__ unsigned short Sb[64][64];    // bf16 shadow, transposed: [d][m]
  __shared__ unsigned short Qc[32][64];    // [t][m]
  __shared__ unsigned short Kc[32][64];    // [t][m]
  __shared__ unsigned short KcT[64][32];   // [m][t]
  __shared__ unsigned short VcT[64][32];   // [d][t]
  __shared__ unsigned short Am[32][32];    // masked scores bf16 [t][t']
  __shared__ float          Zc[32][64];    // in-chunk cumulative K sums
  __shared__ float          zrow[64];
  __shared__ float          denom[32];

  const int tid  = threadIdx.x;
  const int lane = tid & 31;
  const int wid  = tid >> 5;
  const int b    = blockIdx.x >> 3;
  const int h    = blockIdx.x & 7;

  const float* Qg = Qf + (size_t)b * L * 512 + h * 64;
  const float* Kg = Kf + (size_t)b * L * 512 + h * 64;
  const float* Vg = V  + (size_t)b * L * 512 + h * 64;
  float*       Og = Out + (size_t)b * L * 512 + h * 64;

#pragma unroll
  for (int e = 0; e < 16; e++) { int i = tid + e * 256; S[i >> 6][i & 63] = 0.f; }
  if (tid < 64) zrow[tid] = 0.f;
  __syncthreads();

  const int rbase = (lane >> 4) << 3;
  const int cl    = lane & 15;

  for (int c0 = 0; c0 < L; c0 += 32) {
#pragma unroll
    for (int e = 0; e < 8; e++) {
      int idx = tid + e * 256;
      int t = idx >> 6, m = idx & 63;
      size_t g = (size_t)(c0 + t) * 512 + m;
      unsigned short qb = f2bf(Qg[g]);
      unsigned short kb = f2bf(Kg[g]);
      unsigned short vb = f2bf(Vg[g]);
      Qc[t][m]  = qb;
      Kc[t][m]  = kb;
      KcT[m][t] = kb;
      VcT[m][t] = vb;
    }
#pragma unroll
    for (int e = 0; e < 16; e++) {
      int idx = tid + e * 256;
      int m = idx >> 6, d = idx & 63;
      Sb[d][m] = f2bf(S[m][d]);
    }
    __syncthreads();

    if (wid < 4) {
      int t0 = (wid >> 1) * 16, u0 = (wid & 1) * 16;
      v8f a = {};
      a = WMMA_BF16(frag_a(&Qc[0][0], 64, t0, 0),  frag_b(&Kc[0][0], 64, u0, 0),  a);
      a = WMMA_BF16(frag_a(&Qc[0][0], 64, t0, 32), frag_b(&Kc[0][0], 64, u0, 32), a);
#pragma unroll
      for (int r = 0; r < 8; r++) {
        int tr = t0 + rbase + r, tc = u0 + cl;
        Am[tr][tc] = f2bf(tc <= tr ? a[r] : 0.f);
      }
    }
    if (tid < 64) {
      int m = tid;
      float cum = zrow[m];
      for (int t = 0; t < 32; t++) {
        cum += Kg[(size_t)(c0 + t) * 512 + m];
        Zc[t][m] = cum;
      }
      zrow[m] = cum;
    }
    __syncthreads();

    if (tid < 32) {
      float sd = 0.f;
      for (int m = 0; m < 64; m++)
        sd = fmaf(Qg[(size_t)(c0 + tid) * 512 + m], Zc[tid][m], sd);
      denom[tid] = sd;
    }
    __syncthreads();

    {
      int t0 = (wid >> 2) * 16, d0 = (wid & 3) * 16;
      v8f n = {};
      n = WMMA_BF16(frag_a(&Am[0][0], 32, t0, 0),  frag_b(&VcT[0][0], 32, d0, 0),  n);
      n = WMMA_BF16(frag_a(&Qc[0][0], 64, t0, 0),  frag_b(&Sb[0][0],  64, d0, 0),  n);
      n = WMMA_BF16(frag_a(&Qc[0][0], 64, t0, 32), frag_b(&Sb[0][0],  64, d0, 32), n);
#pragma unroll
      for (int r = 0; r < 8; r++) {
        int tr = t0 + rbase + r;
        float zi = __builtin_amdgcn_rcpf(denom[tr] + 1e-6f);
        Og[(size_t)(c0 + tr) * 512 + d0 + cl] = n[r] * zi;
      }
    }
    __syncthreads();

#pragma unroll
    for (int s2 = 0; s2 < 2; s2++) {
      int ti = wid + s2 * 8;
      int m0 = (ti >> 2) * 16, d0 = (ti & 3) * 16;
      v8f cacc;
#pragma unroll
      for (int r = 0; r < 8; r++) cacc[r] = S[m0 + rbase + r][d0 + cl];
      cacc = WMMA_BF16(frag_a(&KcT[0][0], 32, m0, 0), frag_b(&VcT[0][0], 32, d0, 0), cacc);
#pragma unroll
      for (int r = 0; r < 8; r++) S[m0 + rbase + r][d0 + cl] = cacc[r];
    }
    __syncthreads();
  }
}

// ---------------------------------------------------------------------------
// LayerNorm over D=512, one block per row.
// ---------------------------------------------------------------------------
__global__ __launch_bounds__(256) void layernorm_kernel(
    const float* __restrict__ in, const float* __restrict__ g,
    const float* __restrict__ bp, float* __restrict__ out)
{
  __shared__ float red[256];
  const int t = threadIdx.x;
  const float* ip = in + (size_t)blockIdx.x * 512;
  float x0 = ip[t], x1 = ip[t + 256];

  red[t] = x0 + x1; __syncthreads();
  for (int o = 128; o > 0; o >>= 1) { if (t < o) red[t] += red[t + o]; __syncthreads(); }
  float mu = red[0] * (1.f / 512.f);
  __syncthreads();

  float d0 = x0 - mu, d1 = x1 - mu;
  red[t] = d0 * d0 + d1 * d1; __syncthreads();
  for (int o = 128; o > 0; o >>= 1) { if (t < o) red[t] += red[t + o]; __syncthreads(); }
  float rstd = rsqrtf(red[0] * (1.f / 512.f) + 1e-5f);

  float* op = out + (size_t)blockIdx.x * 512;
  op[t]       = d0 * rstd * g[t]       + bp[t];
  op[t + 256] = d1 * rstd * g[t + 256] + bp[t + 256];
}

__global__ void cvt_bf16_kernel(const float* __restrict__ in,
                                unsigned short* __restrict__ out, int n) {
  int i = blockIdx.x * 256 + threadIdx.x;
  if (i < n) out[i] = f2bf(in[i]);
}

// W [K,N] f32 -> out [N,K] bf16 (transposed, for contiguous B-tile staging).
__global__ void cvt_wT_kernel(const float* __restrict__ W,
                              unsigned short* __restrict__ out, int K, int N) {
  int i = blockIdx.x * 256 + threadIdx.x;
  if (i < N * K) {
    int n = i / K, k = i - n * K;
    out[i] = f2bf(W[(size_t)k * N + n]);
  }
}

// ---------------------------------------------------------------------------
// Host orchestration
// ---------------------------------------------------------------------------
extern "C" void kernel_launch(void* const* d_in, const int* in_sizes, int n_in,
                              void* d_out, int out_size, void* d_ws, size_t ws_size,
                              hipStream_t stream)
{
  (void)in_sizes; (void)n_in; (void)out_size; (void)ws_size;
  const int B = 2, L = 2048, D = 512, FF = 2048, TOK = B * L, depth = 2;

  const float* x     = (const float*)d_in[0];
  const float* Wq    = (const float*)d_in[1];
  const float* bq    = (const float*)d_in[2];
  const float* Wk    = (const float*)d_in[3];
  const float* bk    = (const float*)d_in[4];
  const float* Wv    = (const float*)d_in[5];
  const float* bv    = (const float*)d_in[6];
  const float* Wo    = (const float*)d_in[7];
  const float* bo    = (const float*)d_in[8];
  const float* omega = (const float*)d_in[9];
  const float* ln1g  = (const float*)d_in[10];
  const float* ln1b  = (const float*)d_in[11];
  const float* W1    = (const float*)d_in[12];
  const float* b1    = (const float*)d_in[13];
  const float* W2    = (const float*)d_in[14];
  const float* b2    = (const float*)d_in[15];
  const float* ln2g  = (const float*)d_in[16];
  const float* ln2b  = (const float*)d_in[17];
  const float* lnfg  = (const float*)d_in[18];
  const float* lnfb  = (const float*)d_in[19];

  char* ws = (char*)d_ws;
  size_t off = 0;
  auto alloc = [&](size_t bytes) -> void* {
    void* p = ws + off; off += (bytes + 255) & ~(size_t)255; return p;
  };
  float* x_cur = (float*)alloc((size_t)TOK * D * 4);
  float* q     = (float*)alloc((size_t)TOK * D * 4);
  float* k     = (float*)alloc((size_t)TOK * D * 4);
  float* v     = (float*)alloc((size_t)TOK * D * 4);
  float* Qfb   = (float*)alloc((size_t)TOK * D * 4);
  float* Kfb   = (float*)alloc((size_t)TOK * D * 4);
  float* attn  = (float*)alloc((size_t)TOK * D * 4);
  float* t1    = (float*)alloc((size_t)TOK * D * 4);
  float* y1    = (float*)alloc((size_t)TOK * FF * 4);
  unsigned short* abf = (unsigned short*)alloc((size_t)TOK * FF * 2);
  unsigned short* wbf = (unsigned short*)alloc((size_t)D * FF * 2);

  hipMemcpyAsync(x_cur, x, (size_t)TOK * D * 4, hipMemcpyDeviceToDevice, stream);

  auto cvt = [&](const float* src, unsigned short* dst, int n) {
    cvt_bf16_kernel<<<dim3((n + 255) / 256), dim3(256), 0, stream>>>(src, dst, n);
  };
  auto cvtT = [&](const float* Wp, unsigned short* dst, int K_, int N_) {
    cvt_wT_kernel<<<dim3((N_ * K_ + 255) / 256), dim3(256), 0, stream>>>(Wp, dst, K_, N_);
  };

  for (int l = 0; l < depth; l++) {
    const size_t wDD = (size_t)l * D * D;
    // ---- Q/K/V projections ----
    cvt(x_cur, abf, TOK * D);
    cvtT(Wq + wDD, wbf, D, D);
    gemm_wmma_kernel<false, false><<<dim3(D / 64, TOK / 128), 256, 0, stream>>>(
        abf, wbf, bq + (size_t)l * D, nullptr, q, TOK, D, D);
    cvtT(Wk + wDD, wbf, D, D);
    gemm_wmma_kernel<false, false><<<dim3(D / 64, TOK / 128), 256, 0, stream>>>(
        abf, wbf, bk + (size_t)l * D, nullptr, k, TOK, D, D);
    cvtT(Wv + wDD, wbf, D, D);
    gemm_wmma_kernel<false, false><<<dim3(D / 64, TOK / 128), 256, 0, stream>>>(
        abf, wbf, bv + (size_t)l * D, nullptr, v, TOK, D, D);

    // ---- FAVOR+ features ----
    const float* om = omega + (size_t)l * 64 * 32;
    favor_kernel<<<dim3(TOK), 256, 0, stream>>>(q, om, Qfb, TOK);
    favor_kernel<<<dim3(TOK), 256, 0, stream>>>(k, om, Kfb, TOK);

    // ---- chunked causal linear attention ----
    causal_attn_kernel<<<dim3(16), 256, 0, stream>>>(Qfb, Kfb, v, attn, L);

    // ---- output projection + residual, then LN1 ----
    cvt(attn, abf, TOK * D);
    cvtT(Wo + wDD, wbf, D, D);
    gemm_wmma_kernel<false, true><<<dim3(D / 64, TOK / 128), 256, 0, stream>>>(
        abf, wbf, bo + (size_t)l * D, x_cur, t1, TOK, D, D);
    layernorm_kernel<<<dim3(TOK), 256, 0, stream>>>(
        t1, ln1g + (size_t)l * D, ln1b + (size_t)l * D, x_cur);

    // ---- FFN: relu(x W1 + b1) W2 + b2 + x, then LN2 ----
    cvt(x_cur, abf, TOK * D);
    cvtT(W1 + (size_t)l * D * FF, wbf, D, FF);
    gemm_wmma_kernel<true, false><<<dim3(FF / 64, TOK / 128), 256, 0, stream>>>(
        abf, wbf, b1 + (size_t)l * FF, nullptr, y1, TOK, FF, D);
    cvt(y1, abf, TOK * FF);
    cvtT(W2 + (size_t)l * FF * D, wbf, FF, D);
    gemm_wmma_kernel<false, true><<<dim3(D / 64, TOK / 128), 256, 0, stream>>>(
        abf, wbf, b2 + (size_t)l * D, x_cur, t1, TOK, D, FF);
    layernorm_kernel<<<dim3(TOK), 256, 0, stream>>>(
        t1, ln2g + (size_t)l * D, ln2b + (size_t)l * D, x_cur);
  }

  layernorm_kernel<<<dim3(TOK), 256, 0, stream>>>(x_cur, lnfg, lnfb, (float*)d_out);
}
